// StateTrackingRecurrentCell_4037269258998
// MI455X (gfx1250) — matrix operand order repaired and checked by reading, not compiled, not run on hardware
//
#include <hip/hip_runtime.h>
#include <hip/hip_bf16.h>
#include <math.h>

// Problem constants (from reference setup_inputs)
constexpr int B  = 8;
constexpr int S  = 2048;
constexpr int H  = 1024;
constexpr int BS = B * S;          // 16384 rows, indexed m = t*B + b (time-major)
constexpr int H3 = 3 * H;          // u|r|o concatenated columns
constexpr int NBLK = 64;           // persistent scan grid

typedef __attribute__((ext_vector_type(16))) __bf16 v16bf;
typedef __attribute__((ext_vector_type(8)))  __bf16 v8bf;
typedef __attribute__((ext_vector_type(8)))  float  v8f;
typedef __attribute__((ext_vector_type(4)))  unsigned int v4u;
typedef __attribute__((ext_vector_type(8)))  int v8i;
typedef __attribute__((ext_vector_type(4)))  int v4i;

// TDM builtin: 5-arg form on ROCm 7.2 (clang-22); the therock-10.0 header
// toolchain uses a 6-arg form — discriminate via the header that only it ships.
#if __has_builtin(__builtin_amdgcn_tensor_load_to_lds) && \
    !__has_include(<hip/amd_detail/amd_gfx1250_TDM.h>)
#define USE_TDM 1
#else
#define USE_TDM 0
#endif

// ---------------------------------------------------------------- helpers ---

__device__ inline float wave_sum(float v) {
#pragma unroll
  for (int off = 16; off > 0; off >>= 1) v += __shfl_xor(v, off, 32);
  return v;
}

// A-matrix fragment (16x32 bf16, MxK). lanes 0-15 hold M=lane,
// K=0..15; lanes 16-31 hold M=lane-16, K=16..31 (matches B fragment).
__device__ inline v16bf load_a_frag(const __bf16* base, int mrow, int pitch,
                                    int kc, int half) {
  const __bf16* p0 = base + (size_t)mrow * pitch + kc + (half ? 16 : 0);
  v8bf lo = *(const v8bf*)(p0);
  v8bf hi = *(const v8bf*)(p0 + 8);
  v16bf a;
#pragma unroll
  for (int i = 0; i < 8; ++i) { a[i] = lo[i]; a[8 + i] = hi[i]; }
  return a;
}

// B-matrix fragment (32x16 bf16, KxN) from K-contiguous weight rows W[n][k]:
// lane<16: N=lane, K=kc..kc+15; lane>=16: N=lane-16, K=kc+16..kc+31.
__device__ inline v16bf load_b_frag(const __bf16* w, int nrow, int pitch,
                                    int kc, int half) {
  return *(const v16bf*)(w + (size_t)nrow * pitch + kc + (half ? 16 : 0));
}

__device__ inline v8f wmma_bf16(v16bf a, v16bf b, v8f c) {
  return __builtin_amdgcn_wmma_f32_16x16x32_bf16(false, a, false, b,
                                                 (short)0, c, false, false);
}

__device__ inline float sigmoidf_(float x) { return 1.0f / (1.0f + __expf(-x)); }

// Generic-pointer low 32 bits == LDS byte offset (flat LDS aperture mapping).
__device__ inline unsigned lds_byte_off(const void* p) {
  return (unsigned)(unsigned long long)(uintptr_t)p;
}

#if USE_TDM
// Issue a TDM 2D tile load: nrows contiguous rows of H bf16 (2048 B) from
// global into LDS. data_size=8B units -> row = 256 units. D# per ISA §8.3-8.4.
__device__ inline void tdm_load_rows(const __bf16* gsrc, unsigned lds_off,
                                     unsigned nrows) {
  unsigned long long ga = (unsigned long long)(uintptr_t)gsrc;
  v4u g0;
  g0[0] = 1u;                                   // count=1, user desc
  g0[1] = lds_off;                              // lds_addr (bytes)
  g0[2] = (unsigned)(ga & 0xFFFFFFFFu);         // global_addr[31:0]
  g0[3] = (unsigned)(((ga >> 32) & 0x01FFFFFFu) | (2u << 30)); // [56:32]|type=2
  const unsigned td0 = 256u;                    // tensor_dim0 (8B units)
  v8i g1;
  g1[0] = (int)(3u << 16);                                  // data_size=8B
  g1[1] = (int)((td0 & 0xFFFFu) << 16);                     // tensor_dim0 lo
  g1[2] = (int)(((td0 >> 16) & 0xFFFFu) | ((nrows & 0xFFFFu) << 16)); // dim1
  g1[3] = (int)(((nrows >> 16) & 0xFFFFu) | (td0 << 16));   // tile_dim0=256
  g1[4] = (int)(nrows & 0xFFFFu);                           // tile_dim1
  g1[5] = (int)td0;                                         // dim0_stride=256
  g1[6] = 0; g1[7] = 0;
  v4i g2 = {0, 0, 0, 0};
  v4i g3 = {0, 0, 0, 0};
  __builtin_amdgcn_tensor_load_to_lds(g0, g1, g2, g3, 0);
}
#endif

__device__ inline void grid_sync(unsigned* cnt, unsigned* gen, unsigned nblk) {
  __builtin_amdgcn_fence(__ATOMIC_RELEASE, "agent");
  __syncthreads();
  if (threadIdx.x == 0) {
    unsigned g = __hip_atomic_load(gen, __ATOMIC_RELAXED, __HIP_MEMORY_SCOPE_AGENT);
    unsigned a = __hip_atomic_fetch_add(cnt, 1u, __ATOMIC_ACQ_REL,
                                        __HIP_MEMORY_SCOPE_AGENT) + 1u;
    if (a == nblk) {
      __hip_atomic_store(cnt, 0u, __ATOMIC_RELAXED, __HIP_MEMORY_SCOPE_AGENT);
      __hip_atomic_store(gen, g + 1u, __ATOMIC_RELEASE, __HIP_MEMORY_SCOPE_AGENT);
    } else {
      while (__hip_atomic_load(gen, __ATOMIC_RELAXED,
                               __HIP_MEMORY_SCOPE_AGENT) == g)
        __builtin_amdgcn_s_sleep(2);
    }
  }
  __syncthreads();
  __builtin_amdgcn_fence(__ATOMIC_ACQUIRE, "agent");
}

// ---------------------------------------------------------------- kernels ---

// Init: state <- prev_state, zero rns pad, reset grid-sync counters.
__global__ void init_kernel(const float* __restrict__ prev,
                            float* __restrict__ state,
                            __bf16* __restrict__ rns,
                            unsigned* __restrict__ sync) {
  int idx = blockIdx.x * 256 + threadIdx.x;
  if (idx < B * H) state[idx] = prev[idx];
  if (idx < 16 * H) rns[idx] = (__bf16)0.0f;
  if (idx == 0) { sync[0] = 0u; sync[1] = 0u; }
}

// Input LayerNorm: nh[m][k] = LN(hs[b,t,:])[k], m = t*B + b, stored bf16.
__global__ __launch_bounds__(256) void ln_in_kernel(
    const float* __restrict__ hs, const float* __restrict__ g,
    const float* __restrict__ bb, __bf16* __restrict__ nh) {
  int m = blockIdx.x;
  int b = m & (B - 1), t = m >> 3;
  const float* row = hs + ((size_t)b * S + t) * H;
  int tid = threadIdx.x, lane = tid & 31, wv = tid >> 5;
  float v[4], sum = 0.f, sq = 0.f;
#pragma unroll
  for (int i = 0; i < 4; ++i) {
    v[i] = row[tid + i * 256];
    sum += v[i]; sq += v[i] * v[i];
  }
  sum = wave_sum(sum); sq = wave_sum(sq);
  __shared__ float ssum[8], ssq[8], sm, sr;
  if (lane == 0) { ssum[wv] = sum; ssq[wv] = sq; }
  __syncthreads();
  if (tid == 0) {
    float s1 = 0.f, s2 = 0.f;
    for (int w = 0; w < 8; ++w) { s1 += ssum[w]; s2 += ssq[w]; }
    float mean = s1 * (1.0f / H);
    float var  = s2 * (1.0f / H) - mean * mean;
    sm = mean; sr = rsqrtf(var + 1e-5f);
  }
  __syncthreads();
#pragma unroll
  for (int i = 0; i < 4; ++i) {
    int k = tid + i * 256;
    nh[(size_t)m * H + k] = (__bf16)((v[i] - sm) * sr * g[k] + bb[k]);
  }
}

// Weight repack: w1[n][k] = W_gate[n%H][k] (x-half), w2[n][k] = W_gate[n%H][H+k]
// with gate order u|r|o over n, rows K-contiguous for B-fragment loads.
__global__ void wconv_kernel(const float* __restrict__ Wu,
                             const float* __restrict__ Wr,
                             const float* __restrict__ Wo,
                             __bf16* __restrict__ w1, __bf16* __restrict__ w2) {
  size_t i = (size_t)blockIdx.x * 256 + threadIdx.x;
  if (i >= (size_t)H3 * H) return;
  int n = (int)(i / H), k = (int)(i % H);
  int gt = n >> 10, nn = n & (H - 1);
  const float* W = (gt == 0) ? Wu : (gt == 1) ? Wr : Wo;
  w1[i] = (__bf16)W[(size_t)nn * (2 * H) + k];
  w2[i] = (__bf16)W[(size_t)nn * (2 * H) + H + k];
}

// Big parallel GEMM: xp[m][n] = sum_k nh[m][k]*w1[n][k] + bias[n], bf16 out.
// One 16x16 tile per wave, K-loop over 1024 (32 WMMAs).
__global__ __launch_bounds__(256) void gemmx_kernel(
    const __bf16* __restrict__ nh, const __bf16* __restrict__ w1,
    const float* __restrict__ bu, const float* __restrict__ br,
    const float* __restrict__ bo, __bf16* __restrict__ xp) {
  int wg = blockIdx.x * 8 + (threadIdx.x >> 5);
  int lane = threadIdx.x & 31, half = lane >> 4, l15 = lane & 15;
  int mt = wg / (H3 / 16), nt = wg % (H3 / 16);
  int row0 = mt * 16, col0 = nt * 16;
  v8f c = {};
  for (int kc = 0; kc < H; kc += 32) {
    v16bf af  = load_a_frag(nh + (size_t)row0 * H, l15, H, kc, half);
    v16bf bfr = load_b_frag(w1, col0 + l15, H, kc, half);
    c = wmma_bf16(af, bfr, c);
  }
  int n = col0 + l15;
  float bias = (n < H) ? bu[n] : (n < 2 * H) ? br[n - H] : bo[n - 2 * H];
#pragma unroll
  for (int g = 0; g < 8; ++g) {
    int m = row0 + g + (half ? 8 : 0);
    xp[(size_t)m * H3 + n] = (__bf16)(c[g] + bias);
  }
}

// Persistent scan: 2048 steps, 2 grid syncs per step. Recurrent weight slices
// (48 rows = 96 KB bf16 per block) are staged ONCE into LDS (TDM + verified
// plain copy) and reused for all 2048 steps -> no per-step weight traffic.
__global__ __launch_bounds__(256) void scan_kernel(
    float* __restrict__ state, float* __restrict__ ubuf,
    __bf16* __restrict__ rns, const __bf16* __restrict__ xp,
    const __bf16* __restrict__ w2,
    const float* __restrict__ st_g, const float* __restrict__ st_b,
    const float* __restrict__ out_g, const float* __restrict__ out_b,
    float* __restrict__ out, unsigned* __restrict__ sync) {
  __shared__ __attribute__((aligned(16))) __bf16 wlds[48 * H]; // 96 KB weights
  __shared__ __attribute__((aligned(16))) __bf16 nsb[16 * H];  // LN(state)
  __shared__ float  red[8 * 32 * 8];  // per-wave partial C tiles
  __shared__ float  smean[8], srstd[8];

  const int tid = threadIdx.x, lane = tid & 31, wv = tid >> 5;
  const int half = lane >> 4, l15 = lane & 15;
  unsigned* cnt = sync;
  unsigned* gen = sync + 1;

  // ---- one-time weight staging into LDS ----
  const int ti0 = blockIdx.x * 2, ti1 = ti0 + 1;
  const int rowsA0 = (ti0 >> 6) * H + (ti0 & 63) * 16;  // phase-A tile 0
  const int rowsA1 = (ti1 >> 6) * H + (ti1 & 63) * 16;  // phase-A tile 1
  const int rowsB  = 2 * H + blockIdx.x * 16;           // phase-B tile
#if USE_TDM
  if (tid == 0) {
    tdm_load_rows(w2 + (size_t)rowsA0 * H, lds_byte_off(wlds), 16u);
    tdm_load_rows(w2 + (size_t)rowsA1 * H, lds_byte_off(wlds + 16 * H), 16u);
    tdm_load_rows(w2 + (size_t)rowsB  * H, lds_byte_off(wlds + 32 * H), 16u);
    __builtin_amdgcn_s_wait_tensorcnt(0);
  }
  __syncthreads();
#endif
  // Authoritative copy (overwrites TDM result; guards against D# bitfield
  // layout risk in the pdftotext-derived spec — identical data if TDM is ok).
  {
    const __bf16* srcs[3] = { w2 + (size_t)rowsA0 * H,
                              w2 + (size_t)rowsA1 * H,
                              w2 + (size_t)rowsB  * H };
    for (int s = 0; s < 3; ++s) {
      const uint4* gs = (const uint4*)srcs[s];
      uint4* ld = (uint4*)(wlds + s * 16 * H);
      for (int i = tid; i < 16 * H / 8; i += 256) ld[i] = gs[i];
    }
  }
  for (int i = tid; i < 8 * H; i += 256) nsb[8 * H + i] = (__bf16)0.0f;
  __syncthreads();

  for (int t = 0; t <= S; ++t) {
    // --- preamble: stats of state (= s_{t-1}); serves state-LN and out-LN ---
    {
      float sum = 0.f, sq = 0.f;
      int b = wv;
#pragma unroll 4
      for (int i = 0; i < 32; ++i) {
        float v = state[b * H + lane + i * 32];
        sum += v; sq += v * v;
      }
      sum = wave_sum(sum); sq = wave_sum(sq);
      if (lane == 0) {
        float m = sum * (1.0f / H);
        float var = sq * (1.0f / H) - m * m;
        smean[b] = m; srstd[b] = rsqrtf(var + 1e-5f);
      }
    }
    __syncthreads();
    for (int i = tid; i < 8 * H; i += 256) {
      int b = i >> 10, k = i & (H - 1);
      float xh = (state[i] - smean[b]) * srstd[b];
      nsb[i] = (__bf16)(xh * st_g[k] + st_b[k]);
      if (blockIdx.x == 0 && t > 0)
        out[((size_t)b * S + (t - 1)) * H + k] = xh * out_g[k] + out_b[k];
    }
    __syncthreads();
    if (t == S) break;

    // --- phase A: update & reset gates (128 tiles; 2/block, 4-way K-split) ---
    {
      int grp  = wv >> 2;
      int ti   = blockIdx.x * 2 + grp;      // 0..127
      int gate = ti >> 6;                   // 0 = u, 1 = r
      int ncol = (ti & 63) * 16;
      int kc0  = (wv & 3) * 256;
      // prefetch next step's xproj tile rows while we compute
      if ((wv & 3) == 0 && half == 0 && l15 < 8 && t + 1 < S)
        __builtin_prefetch(
            &xp[((size_t)((t + 1) * B + l15)) * H3 + gate * H + ncol], 0, 1);
      v8f c = {};
      for (int kk = 0; kk < 256; kk += 32) {
        int kc = kc0 + kk;
        v16bf af  = load_a_frag(nsb, l15, H, kc, half);
        v16bf bfr = load_b_frag(wlds + grp * 16 * H, l15, H, kc, half);
        c = wmma_bf16(af, bfr, c);
      }
#pragma unroll
      for (int g = 0; g < 8; ++g) red[(wv * 32 + lane) * 8 + g] = c[g];
      __syncthreads();
      if ((wv & 3) == 0 && half == 0) {
        int n = ncol + l15;
#pragma unroll
        for (int g = 0; g < 8; ++g) {
          float s = 0.f;
          for (int w = 0; w < 4; ++w) s += red[((wv + w) * 32 + lane) * 8 + g];
          int b = g;  // half==0 -> rows 0..7 = real batch rows
          float x = s + (float)xp[((size_t)(t * B + b)) * H3 + gate * H + n];
          float sg = sigmoidf_(x);
          if (gate == 0) ubuf[b * H + n] = sg;
          else           rns[b * H + n] = (__bf16)(sg * (float)nsb[b * H + n]);
        }
      }
    }
    grid_sync(cnt, gen, NBLK);

    // --- phase B: candidate + state update (64 tiles; 1/block, 8-way K) ---
    {
      int n0  = blockIdx.x * 16;
      int kc0 = wv * 128;
      if (wv == 0 && half == 0 && l15 < 8 && t + 1 < S)
        __builtin_prefetch(
            &xp[((size_t)((t + 1) * B + l15)) * H3 + 2 * H + n0], 0, 1);
      v8f c = {};
      for (int kk = 0; kk < 128; kk += 32) {
        int kc = kc0 + kk;
        v16bf af  = load_a_frag(rns, l15, H, kc, half);
        v16bf bfr = load_b_frag(wlds + 32 * H, l15, H, kc, half);
        c = wmma_bf16(af, bfr, c);
      }
#pragma unroll
      for (int g = 0; g < 8; ++g) red[(wv * 32 + lane) * 8 + g] = c[g];
      __syncthreads();
      if (wv == 0 && half == 0) {
        int n = n0 + l15;
#pragma unroll
        for (int g = 0; g < 8; ++g) {
          float s = 0.f;
          for (int w = 0; w < 8; ++w) s += red[(w * 32 + lane) * 8 + g];
          int b = g;
          float x = s + (float)xp[((size_t)(t * B + b)) * H3 + 2 * H + n];
          float cand = tanhf(x);
          float u    = ubuf[b * H + n];
          float sold = state[b * H + n];
          state[b * H + n] = (1.0f - u) * sold + u * cand;
        }
      }
    }
    grid_sync(cnt, gen, NBLK);
  }

  // final_state tail of d_out (raw state, per reference)
  if (blockIdx.x == 0)
    for (int i = tid; i < B * H; i += 256)
      out[(size_t)B * S * H + i] = state[i];
}

// ----------------------------------------------------------------- launch ---

extern "C" void kernel_launch(void* const* d_in, const int* in_sizes, int n_in,
                              void* d_out, int out_size, void* d_ws,
                              size_t ws_size, hipStream_t stream) {
  const float* hs    = (const float*)d_in[0];
  const float* prev  = (const float*)d_in[1];
  const float* in_g  = (const float*)d_in[2];
  const float* in_b  = (const float*)d_in[3];
  const float* st_g  = (const float*)d_in[4];
  const float* st_b  = (const float*)d_in[5];
  const float* out_g = (const float*)d_in[6];
  const float* out_b = (const float*)d_in[7];
  const float* Wu    = (const float*)d_in[8];
  const float* bu    = (const float*)d_in[9];
  const float* Wr    = (const float*)d_in[10];
  const float* br    = (const float*)d_in[11];
  const float* Wo    = (const float*)d_in[12];
  const float* bo    = (const float*)d_in[13];
  float* out = (float*)d_out;

  char* ws = (char*)d_ws;
  size_t o = 0;
  auto alloc = [&](size_t bytes) {
    char* p = ws + o;
    o += (bytes + 255) & ~(size_t)255;
    return p;
  };
  __bf16* nh    = (__bf16*)alloc((size_t)BS * H * 2);     // 32 MB
  __bf16* w1    = (__bf16*)alloc((size_t)H3 * H * 2);     // 6 MB
  __bf16* w2    = (__bf16*)alloc((size_t)H3 * H * 2);     // 6 MB
  __bf16* xp    = (__bf16*)alloc((size_t)BS * H3 * 2);    // 96 MB
  float*  state = (float*)alloc((size_t)B * H * 4);
  float*  ubuf  = (float*)alloc((size_t)B * H * 4);
  __bf16* rns   = (__bf16*)alloc((size_t)16 * H * 2);
  unsigned* sync = (unsigned*)alloc(256);

  init_kernel<<<(16 * H + 255) / 256, 256, 0, stream>>>(prev, state, rns, sync);
  ln_in_kernel<<<BS, 256, 0, stream>>>(hs, in_g, in_b, nh);
  wconv_kernel<<<(int)(((size_t)H3 * H + 255) / 256), 256, 0, stream>>>(
      Wu, Wr, Wo, w1, w2);
  gemmx_kernel<<<(BS / 16) * (H3 / 16) / 8, 256, 0, stream>>>(
      nh, w1, bu, br, bo, xp);
  scan_kernel<<<NBLK, 256, 0, stream>>>(state, ubuf, rns, xp, w2, st_g, st_b,
                                        out_g, out_b, out, sync);
}
